// TransformerLayerShard_83296595739552
// MI455X (gfx1250) — matrix operand order repaired and checked by reading, not compiled, not run on hardware
//
#include <hip/hip_runtime.h>
#include <hip/hip_bf16.h>

typedef __bf16 bf16;
typedef __attribute__((ext_vector_type(16))) __bf16 v16bf;
typedef __attribute__((ext_vector_type(8)))  __bf16 v8bf;
typedef __attribute__((ext_vector_type(8)))  float  v8f;

#define SEQ   2048
#define DM    4096
#define NH    16
#define DH    256
#define DFF   16384
#define LDSW  40   // padded row stride (bf16 elems) -> bank-conflict-free frag reads
#define BM    128
#define BN    256

// ---------------------------------------------------------------- LayerNorm
__global__ __launch_bounds__(256)
void ln_kernel(const float* __restrict__ x, const float* __restrict__ sc,
               const float* __restrict__ of, bf16* __restrict__ h) {
  int row = blockIdx.x, tid = threadIdx.x;
  const float* xr = x + (long)row * DM;
  __shared__ float ls[256], ls2[256];
  float s = 0.f, s2 = 0.f;
  for (int i = tid; i < DM; i += 256) { float v = xr[i]; s += v; s2 += v * v; }
  ls[tid] = s; ls2[tid] = s2; __syncthreads();
  for (int st = 128; st > 0; st >>= 1) {
    if (tid < st) { ls[tid] += ls[tid + st]; ls2[tid] += ls2[tid + st]; }
    __syncthreads();
  }
  float mean = ls[0] * (1.0f / DM);
  float var  = ls2[0] * (1.0f / DM) - mean * mean;
  float r    = rsqrtf(var + 1e-5f);
  for (int i = tid; i < DM; i += 256) {
    float v = (xr[i] - mean) * r * sc[i] + of[i];
    h[(long)row * DM + i] = (bf16)v;
  }
}

// ------------------------------------------- fp32 W[K][N] -> bf16 Wt[N][K]
__global__ __launch_bounds__(256)
void transpose_bf16(const float* __restrict__ in, bf16* __restrict__ out,
                    int K, int N) {
  __shared__ float t[32][33];
  int nb = blockIdx.x * 32, kb = blockIdx.y * 32;
  int tx = threadIdx.x, ty = threadIdx.y;          // 32 x 8
  for (int r = ty; r < 32; r += 8) t[r][tx] = in[(long)(kb + r) * N + nb + tx];
  __syncthreads();
  for (int r = ty; r < 32; r += 8)
    out[(long)(nb + r) * K + kb + tx] = (bf16)t[tx][r];
}

// ---------------------------------------------------------------- Softmax
__global__ __launch_bounds__(256)
void softmax_kernel(bf16* __restrict__ logits) {
  long row = blockIdx.x;
  bf16* p = logits + row * (long)SEQ;
  __shared__ float buf[SEQ];
  __shared__ float red[256];
  int tid = threadIdx.x;
  float mx = -3.0e38f;
  for (int i = tid; i < SEQ; i += 256) { float v = (float)p[i]; buf[i] = v; mx = fmaxf(mx, v); }
  red[tid] = mx; __syncthreads();
  for (int st = 128; st > 0; st >>= 1) {
    if (tid < st) red[tid] = fmaxf(red[tid], red[tid + st]);
    __syncthreads();
  }
  mx = red[0]; __syncthreads();
  float s = 0.f;
  for (int i = tid; i < SEQ; i += 256) { float e = __expf(buf[i] - mx); buf[i] = e; s += e; }
  red[tid] = s; __syncthreads();
  for (int st = 128; st > 0; st >>= 1) {
    if (tid < st) red[tid] += red[tid + st];
    __syncthreads();
  }
  float inv = 1.0f / red[0];
  for (int i = tid; i < SEQ; i += 256) p[i] = (bf16)(buf[i] * inv);
}

__device__ __forceinline__ float gelu_tanh(float x) {
  float x3 = x * x * x;
  return 0.5f * x * (1.0f + tanhf(0.7978845608028654f * (x + 0.044715f * x3)));
}

// per-lane async 16B global -> LDS copy, tracked by ASYNCcnt
__device__ __forceinline__ void async_ld16(unsigned lds, const bf16* g) {
  asm volatile("global_load_async_to_lds_b128 %0, %1, off"
               :: "v"(lds), "v"(g) : "memory");
}
__device__ __forceinline__ void wait_async0() {
  asm volatile("s_wait_asynccnt 0x0" ::: "memory");
}

// ---------------------------------------------------------------- WMMA GEMM
// C[M,N] = A[M,K] (bf16 row-major, lda) * Bt[N,K]^T (bf16 row-major, ldb)
// 128x256 block tile, 8 waves (2x4), each wave a 64x64 region (4x4 WMMA
// tiles), K-step 32, double-buffered LDS fed by GLOBAL_LOAD_ASYNC_TO_LDS_B128.
// EPI: 1 logits(scale+bias+causal->bf16)  2 bf16 store (+z col offset)
//      3 bf16 transposed store            4 bias+gelu->bf16
//      5 f32 store                        6 f32 += acc + bias
template <int EPI>
__global__ __launch_bounds__(256)
void gemm_wmma(const bf16* __restrict__ A, long lda, long aOffZ,
               const bf16* __restrict__ B, long ldb, long bOffZ,
               bf16* __restrict__ outB, float* __restrict__ outF,
               long ldc, long cOffZ,
               const float* __restrict__ bias, float scale, int K) {
  const int z  = blockIdx.z;
  const bf16* Ab = A + (long)z * aOffZ;
  const bf16* Bb = B + (long)z * bOffZ;
  const int m0 = blockIdx.y * BM, n0 = blockIdx.x * BN;
  const int tid = threadIdx.x, lane = tid & 31, wave = tid >> 5;
  const int wm = wave >> 2, wn = wave & 3;       // 2 x 4 wave grid, 64x64 each
  const int hf = lane >> 4, l15 = lane & 15;

  __shared__ bf16 As[2][BM * LDSW];
  __shared__ bf16 Bs[2][BN * LDSW];
  v8f acc[4][4] = {};

  // per thread: 2 16B chunks of the A tile (128x32), 4 of the B tile (256x32)
  const int row0 = tid >> 2;            // 0..63
  const int col  = (tid & 3) * 8;
  const bf16* aSrc0 = Ab + (long)(m0 + row0)       * lda + col;
  const bf16* aSrc1 = Ab + (long)(m0 + row0 + 64)  * lda + col;
  const bf16* bSrc0 = Bb + (long)(n0 + row0)       * ldb + col;
  const bf16* bSrc1 = Bb + (long)(n0 + row0 + 64)  * ldb + col;
  const bf16* bSrc2 = Bb + (long)(n0 + row0 + 128) * ldb + col;
  const bf16* bSrc3 = Bb + (long)(n0 + row0 + 192) * ldb + col;
  const int nk = K >> 5;

  const unsigned ldsA0 = (unsigned)(uintptr_t)&As[0][row0 * LDSW + col];
  const unsigned ldsA1 = ldsA0 + 64 * LDSW * 2;
  const unsigned ldsB0 = (unsigned)(uintptr_t)&Bs[0][row0 * LDSW + col];
  const unsigned ldsB1 = ldsB0 + 64 * LDSW * 2;
  const unsigned ldsB2 = ldsB0 + 128 * LDSW * 2;
  const unsigned ldsB3 = ldsB0 + 192 * LDSW * 2;
  const unsigned bufA  = BM * LDSW * (unsigned)sizeof(bf16);
  const unsigned bufBy = BN * LDSW * (unsigned)sizeof(bf16);

  // prologue: stage 0 in flight
  async_ld16(ldsA0, aSrc0); async_ld16(ldsA1, aSrc1);
  async_ld16(ldsB0, bSrc0); async_ld16(ldsB1, bSrc1);
  async_ld16(ldsB2, bSrc2); async_ld16(ldsB3, bSrc3);

  for (int i = 0; i < nk; ++i) {
    const int cur = i & 1;
    wait_async0();          // stage i landed (stage i+1 not yet issued)
    __syncthreads();        // all waves' data visible; prev compute on other buf done
    if (i + 1 < nk) {
      const unsigned oa = (cur ^ 1) * bufA, ob = (cur ^ 1) * bufBy;
      const int kn = (i + 1) * 32;
      async_ld16(ldsA0 + oa, aSrc0 + kn); async_ld16(ldsA1 + oa, aSrc1 + kn);
      async_ld16(ldsB0 + ob, bSrc0 + kn); async_ld16(ldsB1 + ob, bSrc1 + kn);
      async_ld16(ldsB2 + ob, bSrc2 + kn); async_ld16(ldsB3 + ob, bSrc3 + kn);
    }

    const bf16* as = As[cur];
    const bf16* bs = Bs[cur];
    // B fragments: two 8-K chunks at {8*hf, 16+8*hf}, matching A
    v16bf bfrag[4];
#pragma unroll
    for (int nt = 0; nt < 4; ++nt) {
      int nl = wn * 64 + nt * 16 + l15;
      v8bf lo = *(const v8bf*)&bs[nl * LDSW + 8 * hf];
      v8bf hi = *(const v8bf*)&bs[nl * LDSW + 16 + 8 * hf];
      bfrag[nt] = __builtin_shufflevector(lo, hi, 0,1,2,3,4,5,6,7,8,9,10,11,12,13,14,15);
    }
#pragma unroll
    for (int mt = 0; mt < 4; ++mt) {
      // A fragment: two 8-K chunks at {8*hf, 16+8*hf} (ISA 7.12.2 layout)
      int ml = wm * 64 + mt * 16 + l15;
      v8bf lo = *(const v8bf*)&as[ml * LDSW + 8 * hf];
      v8bf hi = *(const v8bf*)&as[ml * LDSW + 16 + 8 * hf];
      v16bf afrag = __builtin_shufflevector(lo, hi, 0,1,2,3,4,5,6,7,8,9,10,11,12,13,14,15);
#pragma unroll
      for (int nt = 0; nt < 4; ++nt)
        acc[mt][nt] = __builtin_amdgcn_wmma_f32_16x16x32_bf16(
            false, afrag, false, bfrag[nt], (short)0, acc[mt][nt], false, false);
    }
  }

  // epilogue: C layout m = i + 8*half (16x16 f32 C table), n = lane&15
#pragma unroll
  for (int mt = 0; mt < 4; ++mt)
#pragma unroll
    for (int nt = 0; nt < 4; ++nt)
#pragma unroll
      for (int i = 0; i < 8; ++i) {
        int m = m0 + wm * 64 + mt * 16 + hf * 8 + i;
        int n = n0 + wn * 64 + nt * 16 + l15;
        float v = acc[mt][nt][i];
        if (EPI == 1) {                         // QK^T: scale + causal + bias
          v = v * scale + bias[(long)m * SEQ + n];
          if (n > m) v -= 1.0e10f;
          outB[(long)z * cOffZ + (long)m * ldc + n] = (bf16)v;
        } else if (EPI == 2) {                  // bf16 row-major (+z col off)
          outB[(long)m * ldc + (long)z * cOffZ + n] = (bf16)v;
        } else if (EPI == 3) {                  // bf16 transposed (V -> Vt)
          outB[(long)n * ldc + m] = (bf16)v;
        } else if (EPI == 4) {                  // FFN1: bias + gelu
          outB[(long)m * ldc + n] = (bf16)gelu_tanh(v + bias[n]);
        } else if (EPI == 5) {                  // WO: f32 store
          outF[(long)m * ldc + n] = v;
        } else {                                // FFN2: residual add + bias
          outF[(long)m * ldc + n] += v + bias[n];
        }
      }
}

// ---------------------------------------------------------------- host side
extern "C" void kernel_launch(void* const* d_in, const int* in_sizes, int n_in,
                              void* d_out, int out_size, void* d_ws, size_t ws_size,
                              hipStream_t stream) {
  const float* x        = (const float*)d_in[0];
  const float* attnBias = (const float*)d_in[1];
  const float* lnScale  = (const float*)d_in[2];
  const float* lnOffset = (const float*)d_in[3];
  const float* wq       = (const float*)d_in[4];
  const float* wk       = (const float*)d_in[5];
  const float* wv       = (const float*)d_in[6];
  const float* wo       = (const float*)d_in[7];
  const float* wffn     = (const float*)d_in[8];
  const float* bffn     = (const float*)d_in[9];
  const float* wffno    = (const float*)d_in[10];
  const float* bffno    = (const float*)d_in[11];
  float* out = (float*)d_out;

  char* ws = (char*)d_ws;
  size_t off = 0;
  auto alloc = [&](size_t bytes) {
    size_t o = off; off += (bytes + 255) & ~(size_t)255; return o;
  };
  bf16* h       = (bf16*)(ws + alloc((size_t)SEQ * DM  * 2));
  bf16* qb      = (bf16*)(ws + alloc((size_t)SEQ * DM  * 2));
  bf16* kb      = (bf16*)(ws + alloc((size_t)SEQ * DM  * 2));
  bf16* vt      = (bf16*)(ws + alloc((size_t)SEQ * DM  * 2));   // [d_global][T]
  bf16* avec    = (bf16*)(ws + alloc((size_t)SEQ * DM  * 2));
  bf16* ff      = (bf16*)(ws + alloc((size_t)SEQ * DFF * 2));
  bf16* wqT     = (bf16*)(ws + alloc((size_t)DM  * DM  * 2));
  bf16* wkT     = (bf16*)(ws + alloc((size_t)DM  * DM  * 2));
  bf16* wvT     = (bf16*)(ws + alloc((size_t)DM  * DM  * 2));
  bf16* woT     = (bf16*)(ws + alloc((size_t)DM  * DM  * 2));
  bf16* wffnT   = (bf16*)(ws + alloc((size_t)DFF * DM  * 2));
  bf16* wffnoT  = (bf16*)(ws + alloc((size_t)DM  * DFF * 2));
  bf16* logitsP = (bf16*)(ws + alloc((size_t)NH * SEQ * SEQ * 2)); // logits, then P
  (void)ws_size; (void)in_sizes; (void)n_in; (void)out_size;

  dim3 tb(32, 8);
  transpose_bf16<<<dim3(DM / 32,  DM / 32),  tb, 0, stream>>>(wq,    wqT,    DM,  DM);
  transpose_bf16<<<dim3(DM / 32,  DM / 32),  tb, 0, stream>>>(wk,    wkT,    DM,  DM);
  transpose_bf16<<<dim3(DM / 32,  DM / 32),  tb, 0, stream>>>(wv,    wvT,    DM,  DM);
  transpose_bf16<<<dim3(DM / 32,  DM / 32),  tb, 0, stream>>>(wo,    woT,    DM,  DM);
  transpose_bf16<<<dim3(DFF / 32, DM / 32),  tb, 0, stream>>>(wffn,  wffnT,  DM,  DFF);
  transpose_bf16<<<dim3(DM / 32,  DFF / 32), tb, 0, stream>>>(wffno, wffnoT, DFF, DM);

  ln_kernel<<<SEQ, 256, 0, stream>>>(x, lnScale, lnOffset, h);

  gemm_wmma<2><<<dim3(DM / BN, SEQ / BM, 1), 256, 0, stream>>>(
      h, DM, 0, wqT, DM, 0, qb, nullptr, DM, 0, nullptr, 0.f, DM);
  gemm_wmma<2><<<dim3(DM / BN, SEQ / BM, 1), 256, 0, stream>>>(
      h, DM, 0, wkT, DM, 0, kb, nullptr, DM, 0, nullptr, 0.f, DM);
  gemm_wmma<3><<<dim3(DM / BN, SEQ / BM, 1), 256, 0, stream>>>(
      h, DM, 0, wvT, DM, 0, vt, nullptr, SEQ, 0, nullptr, 0.f, DM);

  gemm_wmma<1><<<dim3(SEQ / BN, SEQ / BM, NH), 256, 0, stream>>>(
      qb, DM, DH, kb, DM, DH, logitsP, nullptr, SEQ, (long)SEQ * SEQ,
      attnBias, 1.0f / 16.0f, DH);

  softmax_kernel<<<NH * SEQ, 256, 0, stream>>>(logitsP);

  gemm_wmma<2><<<dim3(DH / BN, SEQ / BM, NH), 256, 0, stream>>>(
      logitsP, SEQ, (long)SEQ * SEQ, vt, SEQ, (long)DH * SEQ,
      avec, nullptr, DM, DH, nullptr, 0.f, SEQ);

  gemm_wmma<5><<<dim3(DM / BN, SEQ / BM, 1), 256, 0, stream>>>(
      avec, DM, 0, woT, DM, 0, nullptr, out, DM, 0, nullptr, 0.f, DM);

  gemm_wmma<4><<<dim3(DFF / BN, SEQ / BM, 1), 256, 0, stream>>>(
      h, DM, 0, wffnT, DM, 0, ff, nullptr, DFF, 0, bffn, 0.f, DM);

  gemm_wmma<6><<<dim3(DM / BN, SEQ / BM, 1), 256, 0, stream>>>(
      ff, DFF, 0, wffnoT, DFF, 0, nullptr, out, DM, 0, bffno, 0.f, DFF);
}
